// RTN_7610682048735
// MI455X (gfx1250) — matrix operand order repaired and checked by reading, not compiled, not run on hardware
//
#include <hip/hip_runtime.h>

// ---------------- problem constants ----------------
#define BATCH 16384
#define IN_CH 4096
#define COMP  512
static constexpr float NOISE_SCALE = 0.22334200f;  // 1/sqrt(2*R*SNR), R=2, SNR=5.01187

// ---------------- vector types for WMMA ----------------
typedef __attribute__((ext_vector_type(8)))  float  v8f;
typedef __attribute__((ext_vector_type(16))) __bf16 v16bf;
typedef __attribute__((ext_vector_type(8)))  __bf16 v8bf;

// ---------------- CDNA5 async global->LDS path ----------------
#if defined(__AMDGCN__) && __has_builtin(__builtin_amdgcn_global_load_async_to_lds_b128)
#define HAVE_ASYNC 1
#else
#define HAVE_ASYNC 0
#endif

#if HAVE_ASYNC
typedef __attribute__((__vector_size__(4 * sizeof(int)))) int v4i_b128;
#define AS_G128(p) ((__attribute__((address_space(1))) v4i_b128*)(p))
#define AS_L128(p) ((__attribute__((address_space(3))) v4i_b128*)(p))
#endif

__device__ __forceinline__ void wait_async_all() {
#if defined(__AMDGCN__)
#if __has_builtin(__builtin_amdgcn_s_wait_asynccnt)
  __builtin_amdgcn_s_wait_asynccnt(0);
#else
  asm volatile("s_wait_asynccnt 0x0" ::: "memory");
#endif
#endif
}

// ---------------- helpers ----------------
__device__ __forceinline__ unsigned short f2bf_u(float f) {
  union { float f; unsigned u; } v; v.f = f;
  unsigned r = v.u + 0x7FFFu + ((v.u >> 16) & 1u);   // round-to-nearest-even
  return (unsigned short)(r >> 16);
}

__device__ __forceinline__ unsigned pcg_hash(unsigned v) {
  unsigned s = v * 747796405u + 2891336453u;
  unsigned w = ((s >> ((s >> 28u) + 4u)) ^ s) * 277803737u;
  return (w >> 22u) ^ w;
}

// deterministic standard normal from flat index (Box-Muller)
__device__ __forceinline__ float gauss_noise(unsigned idx) {
  unsigned h1 = pcg_hash(idx);
  unsigned h2 = pcg_hash(idx ^ 0x9E3779B9u);
  float u1 = ((h1 >> 8) + 1u) * (1.0f / 16777216.0f);   // (0,1]
  float u2 = (h2 >> 8) * (1.0f / 16777216.0f);          // [0,1)
  float r  = __fsqrt_rn(-2.0f * __logf(u1));
  return r * __cosf(6.28318530718f * u2);
}

// ---------------- f32 -> bf16 pack kernel (activations) ----------------
__global__ __launch_bounds__(256) void cvt_f32_bf16(const float* __restrict__ in,
                                                    unsigned short* __restrict__ out,
                                                    int n4) {
  int i = blockIdx.x * blockDim.x + threadIdx.x;
  if (i >= n4) return;
  float4 f = reinterpret_cast<const float4*>(in)[i];
  ushort4 o;
  o.x = f2bf_u(f.x); o.y = f2bf_u(f.y); o.z = f2bf_u(f.z); o.w = f2bf_u(f.w);
  reinterpret_cast<ushort4*>(out)[i] = o;
}

// ---------------- f32 [K,N] -> bf16 transposed [N,K] (weights, once per launch) ----
__global__ __launch_bounds__(256) void cvt_transpose_bf16(const float* __restrict__ in,
                                                          unsigned short* __restrict__ out,
                                                          int K, int N) {
  __shared__ float tile[32][33];
  const int kb = blockIdx.x * 32;
  const int nb = blockIdx.y * 32;
  const int tx = threadIdx.x;   // 0..31
  const int ty = threadIdx.y;   // 0..7
#pragma unroll
  for (int i = 0; i < 32; i += 8)
    tile[ty + i][tx] = in[(size_t)(kb + ty + i) * N + nb + tx];
  __syncthreads();
#pragma unroll
  for (int i = 0; i < 32; i += 8)
    out[(size_t)(nb + ty + i) * K + kb + tx] = f2bf_u(tile[tx][ty + i]);
}

// ---------------- fused bf16 WMMA GEMM (Bt is pre-transposed, [N,K]) ----------------
// C[M,N] = epilogue(A[M,K] @ Bt[N,K]^T + bias)
// EPI: 0 = bias only (f32 out), 1 = bias+relu (bf16 out),
//      2 = bias+clamp[0,1]+AWGN noise (bf16 out)
template <int EPI, typename OutT>
__global__ __launch_bounds__(256) void gemm_bf16_wmma(
    const __bf16* __restrict__ A,
    const __bf16* __restrict__ Bt,
    const float*  __restrict__ bias,
    OutT* __restrict__ C,
    int M, int N, int K) {
  (void)M;
  constexpr int BM = 128, BN = 128, BK = 64;
  constexpr int LDA = BK + 8;    // 72 bf16 = 144B row stride (16B aligned)
  constexpr int LDB = BK + 16;   // 80 bf16 = 160B row stride (32B aligned)
  __shared__ alignas(32) __bf16 As[2][BM][LDA];
  __shared__ alignas(32) __bf16 Bs[2][BN][LDB];

  const int tid  = threadIdx.x;
  const int lane = tid & 31;
  const int wid  = tid >> 5;     // 8 waves
  const int wm   = wid & 1;      // 2 wave rows * 64
  const int wn   = wid >> 1;     // 4 wave cols * 32
  const int l    = lane & 15;
  const int hs   = lane >> 4;

  const int bm = blockIdx.x * BM;
  const int bn = blockIdx.y * BN;

  v8f acc[4][2];
  v8f zero = {0.f, 0.f, 0.f, 0.f, 0.f, 0.f, 0.f, 0.f};
#pragma unroll
  for (int i = 0; i < 4; ++i)
#pragma unroll
    for (int j = 0; j < 2; ++j) acc[i][j] = zero;

  // stage one 128x64 K-contiguous tile pair into LDS buffer `buf`
  auto stage = [&](int buf, int k0) {
#pragma unroll
    for (int i = 0; i < 4; ++i) {
      int idx = tid + i * 256;              // 0..1023
      int r   = idx >> 3;                   // 0..127
      int c   = (idx & 7) * 8;              // 0..56 step 8 (16B chunks)
      const __bf16* ga = &A[(size_t)(bm + r) * K + k0 + c];
      const __bf16* gb = &Bt[(size_t)(bn + r) * K + k0 + c];
      __bf16* la = &As[buf][r][c];
      __bf16* lb = &Bs[buf][r][c];
#if HAVE_ASYNC
      __builtin_amdgcn_global_load_async_to_lds_b128(AS_G128(ga), AS_L128(la), 0, 0);
      __builtin_amdgcn_global_load_async_to_lds_b128(AS_G128(gb), AS_L128(lb), 0, 0);
#else
      *reinterpret_cast<uint4*>(la) = *reinterpret_cast<const uint4*>(ga);
      *reinterpret_cast<uint4*>(lb) = *reinterpret_cast<const uint4*>(gb);
#endif
    }
  };

  const int nk = K / BK;
  stage(0, 0);
  wait_async_all();
  __syncthreads();

  for (int ks = 0; ks < nk; ++ks) {
    const int buf = ks & 1;
    const int next = ks + 1;
    if (next < nk) stage(buf ^ 1, next * BK);          // async fill of other buffer
    if (next + 1 < nk) {                               // L2 prefetch two stages ahead
      __builtin_prefetch(&A[(size_t)(bm + (tid >> 1)) * K + (next + 1) * BK + (tid & 1) * 32], 0, 1);
      __builtin_prefetch(&Bt[(size_t)(bn + (tid >> 1)) * K + (next + 1) * BK + (tid & 1) * 32], 0, 1);
    }

    // compute on `buf` (two 16x16x32 k-steps)
#pragma unroll
    for (int kk = 0; kk < BK; kk += 32) {
      v16bf afrag[4];
#pragma unroll
      for (int tm = 0; tm < 4; ++tm) {
        const __bf16* ap = &As[buf][wm * 64 + tm * 16 + l][kk + hs * 16];
        v8bf alo = *reinterpret_cast<const v8bf*>(ap);
        v8bf ahi = *reinterpret_cast<const v8bf*>(ap + 8);
        afrag[tm] = __builtin_shufflevector(alo, ahi,
                                            0, 1, 2, 3, 4, 5, 6, 7,
                                            8, 9, 10, 11, 12, 13, 14, 15);
      }
      v16bf bfrag[2];
#pragma unroll
      for (int tn = 0; tn < 2; ++tn)
        bfrag[tn] = *reinterpret_cast<const v16bf*>(&Bs[buf][wn * 32 + tn * 16 + l][kk + hs * 16]);

#pragma unroll
      for (int tm = 0; tm < 4; ++tm)
#pragma unroll
        for (int tn = 0; tn < 2; ++tn)
          acc[tm][tn] = __builtin_amdgcn_wmma_f32_16x16x32_bf16(
              false, afrag[tm], false, bfrag[tn], (short)0, acc[tm][tn], false, false);
    }

    wait_async_all();      // our async fills of buf^1 have landed
    __syncthreads();       // everyone done reading buf & filling buf^1
  }

  // ---------------- fused epilogue ----------------
#pragma unroll
  for (int tm = 0; tm < 4; ++tm) {
#pragma unroll
    for (int tn = 0; tn < 2; ++tn) {
      int col = bn + wn * 32 + tn * 16 + l;
      float bv = bias[col];
#pragma unroll
      for (int r = 0; r < 8; ++r) {
        int row = bm + wm * 64 + tm * 16 + hs * 8 + r;
        float v = acc[tm][tn][r] + bv;
        if (EPI == 1) v = fmaxf(v, 0.0f);
        if (EPI == 2) {
          v = fminf(fmaxf(v, 0.0f), 1.0f);
          v += NOISE_SCALE * gauss_noise((unsigned)(row * N + col));
        }
        size_t o = (size_t)row * N + col;
        if constexpr (sizeof(OutT) == 2) {
          union { unsigned short s; OutT b; } cv;
          cv.s = f2bf_u(v);
          C[o] = cv.b;
        } else {
          C[o] = v;
        }
      }
    }
  }
}

// ---------------- launcher ----------------
extern "C" void kernel_launch(void* const* d_in, const int* in_sizes, int n_in,
                              void* d_out, int out_size, void* d_ws, size_t ws_size,
                              hipStream_t stream) {
  (void)in_sizes; (void)n_in; (void)out_size; (void)ws_size;
  const float* x      = (const float*)d_in[0];
  const float* enc_w1 = (const float*)d_in[1];
  const float* enc_b1 = (const float*)d_in[2];
  const float* enc_w2 = (const float*)d_in[3];
  const float* enc_b2 = (const float*)d_in[4];
  const float* dec_w1 = (const float*)d_in[5];
  const float* dec_b1 = (const float*)d_in[6];
  const float* dec_w2 = (const float*)d_in[7];
  const float* dec_b2 = (const float*)d_in[8];

  char* ws = (char*)d_ws;
  size_t off = 0;
  auto take = [&](size_t elems) {
    void* p = ws + off;
    off += (elems * 2 + 255) & ~(size_t)255;
    return (unsigned short*)p;
  };
  unsigned short* xbf  = take((size_t)BATCH * IN_CH);
  unsigned short* w1t  = take((size_t)IN_CH * IN_CH);   // [N,K] = [4096,4096]
  unsigned short* w2t  = take((size_t)IN_CH * COMP);    // [N,K] = [512,4096]
  unsigned short* w3t  = take((size_t)COMP  * IN_CH);   // [N,K] = [4096,512]
  unsigned short* w4t  = take((size_t)IN_CH * IN_CH);   // [N,K] = [4096,4096]
  unsigned short* hbf  = take((size_t)BATCH * IN_CH);
  unsigned short* zbf  = take((size_t)BATCH * COMP);
  unsigned short* h2bf = xbf;   // x consumed after GEMM1 -> reuse its slot for H2

  // activations: straight convert
  {
    int n4 = (int)((size_t)BATCH * IN_CH / 4);
    cvt_f32_bf16<<<dim3((n4 + 255) / 256), dim3(256), 0, stream>>>(x, xbf, n4);
  }
  // weights: convert + transpose to [N,K] so GEMM staging is K-contiguous
  dim3 tblk(32, 8);
  cvt_transpose_bf16<<<dim3(IN_CH / 32, IN_CH / 32), tblk, 0, stream>>>(enc_w1, w1t, IN_CH, IN_CH);
  cvt_transpose_bf16<<<dim3(IN_CH / 32, COMP  / 32), tblk, 0, stream>>>(enc_w2, w2t, IN_CH, COMP);
  cvt_transpose_bf16<<<dim3(COMP  / 32, IN_CH / 32), tblk, 0, stream>>>(dec_w1, w3t, COMP,  IN_CH);
  cvt_transpose_bf16<<<dim3(IN_CH / 32, IN_CH / 32), tblk, 0, stream>>>(dec_w2, w4t, IN_CH, IN_CH);

  dim3 blk(256);
  dim3 gBig(BATCH / 128, IN_CH / 128);   // 128 x 32 blocks
  dim3 gZ(BATCH / 128, COMP / 128);      // 128 x 4 blocks

  // H = relu(X @ W1 + b1)
  gemm_bf16_wmma<1, __bf16><<<gBig, blk, 0, stream>>>(
      (const __bf16*)xbf, (const __bf16*)w1t, enc_b1, (__bf16*)hbf,
      BATCH, IN_CH, IN_CH);
  // Z = clip(H @ W2 + b2, 0, 1) + noise
  gemm_bf16_wmma<2, __bf16><<<gZ, blk, 0, stream>>>(
      (const __bf16*)hbf, (const __bf16*)w2t, enc_b2, (__bf16*)zbf,
      BATCH, COMP, IN_CH);
  // H2 = relu(Z @ W3 + b1)
  gemm_bf16_wmma<1, __bf16><<<gBig, blk, 0, stream>>>(
      (const __bf16*)zbf, (const __bf16*)w3t, dec_b1, (__bf16*)h2bf,
      BATCH, IN_CH, COMP);
  // OUT = H2 @ W4 + b2   (f32 output)
  gemm_bf16_wmma<0, float><<<gBig, blk, 0, stream>>>(
      (const __bf16*)h2bf, (const __bf16*)w4t, dec_b2, (float*)d_out,
      BATCH, IN_CH, IN_CH);
}
